// MultiHeadAttention_73091753443647
// MI455X (gfx1250) — compile-verified, matrix operand order checked by reading, not run on hardware
//
#include <hip/hip_runtime.h>
#include <hip/hip_bf16.h>
#include <stdint.h>

#define D_MODEL 1024
#define NUM_HEADS 16
#define DEPTH 64
#define BATCH 4
#define SEQ 2048
#define MTOT (BATCH * SEQ)   // 8192

typedef __attribute__((ext_vector_type(16))) __bf16 bf16x16;
typedef __attribute__((ext_vector_type(8)))  __bf16 bf16x8;
typedef __attribute__((ext_vector_type(8)))  float  f32x8;
typedef __attribute__((ext_vector_type(8)))  unsigned u32x8;

// ---- bf16 conversion helpers (prefer v_cvt_pk_bf16_f32) ----
static __device__ __forceinline__ unsigned cvt_pk_bf16(float a, float b) {
#if __has_builtin(__builtin_amdgcn_cvt_pk_bf16_f32)
    typedef __attribute__((ext_vector_type(2))) __bf16 bf16x2;
    bf16x2 r = __builtin_amdgcn_cvt_pk_bf16_f32(a, b);
    return __builtin_bit_cast(unsigned, r);
#else
    unsigned ua = __builtin_bit_cast(unsigned, a) + 0x8000u;
    unsigned ub = __builtin_bit_cast(unsigned, b) + 0x8000u;
    return __builtin_amdgcn_perm(ub, ua, 0x07060302);   // {hi16(b),hi16(a)}
#endif
}
static __device__ __forceinline__ __bf16 f2b(float f) {
    unsigned short h = (unsigned short)((__builtin_bit_cast(unsigned, f) + 0x8000u) >> 16);
    return __builtin_bit_cast(__bf16, h);
}
static __device__ __forceinline__ f32x8 fzero() {
    f32x8 v;
#pragma unroll
    for (int i = 0; i < 8; ++i) v[i] = 0.f;
    return v;
}
// xor-16 butterfly across the wave32 halves, VALU-only when possible
static __device__ __forceinline__ float xor16(float x) {
#if __has_builtin(__builtin_amdgcn_permlanex16)
    int i = __builtin_bit_cast(int, x);
    int r = __builtin_amdgcn_permlanex16(i, i, 0x76543210, (int)0xfedcba98u,
                                         false, false);
    return __builtin_bit_cast(float, r);
#else
    return __shfl_xor(x, 16);
#endif
}

// ---- CDNA5 async copy: global -> LDS, 16B/lane, tracked by ASYNCcnt ----
static __device__ __forceinline__ void async_b128(unsigned ldsOff, const void* g) {
    asm volatile("global_load_async_to_lds_b128 %0, %1, off"
                 :: "v"(ldsOff), "v"((unsigned long long)(uintptr_t)g)
                 : "memory");
}
template <int N>
static __device__ __forceinline__ void wait_async_n() {
#if __has_builtin(__builtin_amdgcn_s_wait_asynccnt)
    __builtin_amdgcn_s_wait_asynccnt(N);
#else
    asm volatile("s_wait_asynccnt %0" :: "i"(N) : "memory");
#endif
}
// low 32 bits of a generic pointer to LDS == LDS byte offset (flat aperture)
static __device__ __forceinline__ unsigned lds_off(const void* p) {
    return (unsigned)(uintptr_t)p;
}

// Fragment loader: A-fragment of row-major [rows][stride] bf16 matrix
// (== B-fragment of its transpose). lane L holds row (L&15), K chunks
// {kh*8..+7} and {16+kh*8..+7}, kh = L>>4 (ISA 7.12.2 16-bit layout).
static __device__ __forceinline__ bf16x16 ldfrag(const __bf16* base, int stride,
                                                 int kOff, int lane) {
    const __bf16* p = base + (size_t)(lane & 15) * stride + kOff + ((lane >> 4) << 3);
    bf16x8 lo = *reinterpret_cast<const bf16x8*>(p);
    bf16x8 hi = *reinterpret_cast<const bf16x8*>(p + 16);
    bf16x16 v;
#pragma unroll
    for (int i = 0; i < 8; ++i) { v[i] = lo[i]; v[8 + i] = hi[i]; }
    return v;
}

static __device__ __forceinline__ f32x8 wmma_bf16(bf16x16 a, bf16x16 b, f32x8 c) {
    return __builtin_amdgcn_wmma_f32_16x16x32_bf16(false, a, false, b,
                                                   (short)0, c, false, false);
}

// ---------------------------------------------------------------------------
// Weight transpose + fp32 -> bf16:  Wt[n][k] = bf16(W[k][n])
// ---------------------------------------------------------------------------
__global__ void wtr_73091753443647_kernel(const float* __restrict__ W,
                                          __bf16* __restrict__ Wt) {
    int idx = blockIdx.x * blockDim.x + threadIdx.x;      // 1M elements
    int n = idx & (D_MODEL - 1);
    int k = idx >> 10;
    Wt[(size_t)n * D_MODEL + k] = f2b(W[(size_t)k * D_MODEL + n]);
}

// ---------------------------------------------------------------------------
// WMMA GEMM: C[8192,1024] = (A[8192,1024] x W + bias) * scaleA
// Double-buffered LDS; B (and A when bf16) staged with async copies.
//  MODE 0: bf16 -> [B,H,S,64]   (Q with scale=1/8, K with scale=1)
//  MODE 1: bf16 -> [B,H,64,S]   (V, transposed)
//  MODE 2: f32  -> [8192,1024]  (output projection)
// ---------------------------------------------------------------------------
template <int MODE, bool ABF>
__global__ __launch_bounds__(256) void gemm_73091753443647_kernel(
    const void* __restrict__ Aptr, const __bf16* __restrict__ Wt,
    const float* __restrict__ bias, void* __restrict__ Cptr, float scaleA) {
    constexpr int LDSW = 40;                 // padded row stride (bf16 elems)
    constexpr int NT = D_MODEL / 32;         // 32 K-tiles
    constexpr int PER_TILE = ABF ? 4 : 2;    // async issues per tile per wave
    __shared__ __bf16 As[2][128 * LDSW];
    __shared__ __bf16 Bs[2][128 * LDSW];

    const int t = threadIdx.x;
    const int lane = t & 31;
    const int wid = t >> 5;
    const int mTile = blockIdx.y * 128;
    const int nTile = blockIdx.x * 128;
    const int mBase = (wid >> 2) * 64;       // wave row block (0/64)
    const int nBase = (wid & 3) * 32;        // wave col block (0..96)

    const int lrow = t >> 1;                 // 0..127
    const int lseg = (t & 1) << 4;           // 0 or 16 elems

    auto stage = [&](int buf, int kk) {
        const unsigned ldsB = lds_off(&Bs[buf][lrow * LDSW + lseg]);
        const __bf16* bsrc = Wt + (size_t)(nTile + lrow) * D_MODEL + kk + lseg;
        async_b128(ldsB, bsrc);
        async_b128(ldsB + 16, bsrc + 8);
        if constexpr (ABF) {
            const unsigned ldsA = lds_off(&As[buf][lrow * LDSW + lseg]);
            const __bf16* src = (const __bf16*)Aptr +
                                (size_t)(mTile + lrow) * D_MODEL + kk + lseg;
            async_b128(ldsA, src);
            async_b128(ldsA + 16, src + 8);
        } else {
            const float* src = (const float*)Aptr +
                               (size_t)(mTile + lrow) * D_MODEL + kk + lseg;
            float4 f0 = reinterpret_cast<const float4*>(src)[0];
            float4 f1 = reinterpret_cast<const float4*>(src)[1];
            float4 f2 = reinterpret_cast<const float4*>(src)[2];
            float4 f3 = reinterpret_cast<const float4*>(src)[3];
            uint4 d0 = make_uint4(cvt_pk_bf16(f0.x, f0.y), cvt_pk_bf16(f0.z, f0.w),
                                  cvt_pk_bf16(f1.x, f1.y), cvt_pk_bf16(f1.z, f1.w));
            uint4 d1 = make_uint4(cvt_pk_bf16(f2.x, f2.y), cvt_pk_bf16(f2.z, f2.w),
                                  cvt_pk_bf16(f3.x, f3.y), cvt_pk_bf16(f3.z, f3.w));
            *reinterpret_cast<uint4*>(&As[buf][lrow * LDSW + lseg])     = d0;
            *reinterpret_cast<uint4*>(&As[buf][lrow * LDSW + lseg + 8]) = d1;
        }
    };

    f32x8 acc[4][2];
#pragma unroll
    for (int a = 0; a < 4; ++a)
#pragma unroll
        for (int b = 0; b < 2; ++b) acc[a][b] = fzero();

    stage(0, 0);
    for (int i = 0; i < NT; ++i) {
        const int cur = i & 1;
        if (i + 1 < NT) {
            stage(cur ^ 1, (i + 1) * 32);    // prefetch next tile behind compute
            wait_async_n<PER_TILE>();        // current tile resident
        } else {
            wait_async_n<0>();
        }
        __syncthreads();

        bf16x16 af[4], bfr[2];
#pragma unroll
        for (int fm = 0; fm < 4; ++fm)
            af[fm] = ldfrag(&As[cur][(mBase + fm * 16) * LDSW], LDSW, 0, lane);
#pragma unroll
        for (int fn = 0; fn < 2; ++fn)
            bfr[fn] = ldfrag(&Bs[cur][(nBase + fn * 16) * LDSW], LDSW, 0, lane);
#pragma unroll
        for (int fm = 0; fm < 4; ++fm)
#pragma unroll
            for (int fn = 0; fn < 2; ++fn)
                acc[fm][fn] = wmma_bf16(af[fm], bfr[fn], acc[fm][fn]);
        __syncthreads();
    }

    // ---- epilogue ----
    const int rowAdd = (lane >> 4) << 3;     // +8 for lanes 16..31
    const int ncol = lane & 15;
#pragma unroll
    for (int fm = 0; fm < 4; ++fm)
#pragma unroll
        for (int fn = 0; fn < 2; ++fn)
#pragma unroll
            for (int r = 0; r < 8; ++r) {
                int m = mTile + mBase + fm * 16 + r + rowAdd;
                int n = nTile + nBase + fn * 16 + ncol;
                float v = (acc[fm][fn][r] + bias[n]) * scaleA;
                if constexpr (MODE == 2) {
                    ((float*)Cptr)[(size_t)m * D_MODEL + n] = v;
                } else {
                    int bb = m >> 11, s = m & (SEQ - 1);
                    int h = n >> 6, d = n & 63;
                    if constexpr (MODE == 0)
                        ((__bf16*)Cptr)[((size_t)(bb * NUM_HEADS + h) * SEQ + s) * DEPTH + d] = f2b(v);
                    else
                        ((__bf16*)Cptr)[((size_t)(bb * NUM_HEADS + h) * DEPTH + d) * SEQ + s] = f2b(v);
                }
            }
}

// ---------------------------------------------------------------------------
// Flash attention (causal). Q,K bf16 [B,H,S,64] (Q pre-scaled by 1/8);
// Vt bf16 [B,H,64,S]; Out bf16 [B,S,H*64].
// K/V chunks double-buffered in LDS via async copies; scores^T layout keeps
// softmax stats per-lane + one permlanex16; P^T repacks in-lane to B-frags.
// ---------------------------------------------------------------------------
__global__ __launch_bounds__(256) void attn_73091753443647_kernel(
    const __bf16* __restrict__ Q, const __bf16* __restrict__ K,
    const __bf16* __restrict__ Vt, __bf16* __restrict__ Out) {
    constexpr int LDSW = 72;                 // padded chunk row (bf16 elems)
    __shared__ __bf16 Ks[2][64 * LDSW];
    __shared__ __bf16 Vs[2][64 * LDSW];

    const int tid = threadIdx.x, lane = tid & 31, wid = tid >> 5;
    const int b = blockIdx.z, h = blockIdx.y;
    const int qTile = blockIdx.x;
    const int qRow0 = qTile * 128 + wid * 16;
    const size_t bh = (size_t)b * NUM_HEADS + h;
    const __bf16* Qb = Q + bh * SEQ * DEPTH;
    const __bf16* Kb = K + bh * SEQ * DEPTH;
    const __bf16* Vb = Vt + bh * DEPTH * SEQ;

    // staging map: 64 rows x 64 bf16, 4 threads/row, 32B (=2 b128) each
    const int srow = tid >> 2;
    const int sseg = (tid & 3) << 4;

    auto stage = [&](int buf, int ks) {
        const unsigned ldsK = lds_off(&Ks[buf][srow * LDSW + sseg]);
        const unsigned ldsV = lds_off(&Vs[buf][srow * LDSW + sseg]);
        const __bf16* kg = Kb + (size_t)(ks + srow) * DEPTH + sseg;
        const __bf16* vg = Vb + (size_t)srow * SEQ + ks + sseg;
        async_b128(ldsK, kg);
        async_b128(ldsK + 16, kg + 8);
        async_b128(ldsV, vg);
        async_b128(ldsV + 16, vg + 8);
    };

    // resident Q fragments (WMMA B operand: B^T = Q row-major)
    bf16x16 qf0 = ldfrag(Qb + (size_t)qRow0 * DEPTH, DEPTH, 0, lane);
    bf16x16 qf1 = ldfrag(Qb + (size_t)qRow0 * DEPTH, DEPTH, 32, lane);

    f32x8 o[4];
#pragma unroll
    for (int j = 0; j < 4; ++j) o[j] = fzero();
    float mI = -1e30f, lI = 0.f;
    const int qcol = qRow0 + (lane & 15);
    const int keyAdd = (lane >> 4) << 3;
    const int nCW = (qRow0 + 15) / 64 + 1;   // this wave's causal chunk count
    const int nFullW = qRow0 >> 6;           // chunks needing no mask
    const int nCblk = 2 * qTile + 2;         // block-uniform staging count

    stage(0, 0);
    for (int c = 0; c < nCblk; ++c) {
        const int cur = c & 1;
        const int ks = c * 64;
        if (c + 1 < nCblk) {
            stage(cur ^ 1, ks + 64);         // prefetch next chunk behind compute
            wait_async_n<4>();
        } else {
            wait_async_n<0>();
        }
        __syncthreads();

        if (c < nCW) {
            // scores^T fragments: st[f] = keys ks+f*16.. x 16 queries
            f32x8 st[4];
#pragma unroll
            for (int f = 0; f < 4; ++f) {
                bf16x16 k0 = ldfrag(&Ks[cur][(f * 16) * LDSW], LDSW, 0, lane);
                bf16x16 k1 = ldfrag(&Ks[cur][(f * 16) * LDSW], LDSW, 32, lane);
                f32x8 z = fzero();
                z = wmma_bf16(k0, qf0, z);
                z = wmma_bf16(k1, qf1, z);
                st[f] = z;
            }
            // causal mask only on diagonal-straddling chunks
            if (c >= nFullW) {
#pragma unroll
                for (int f = 0; f < 4; ++f)
#pragma unroll
                    for (int r = 0; r < 8; ++r) {
                        int key = ks + f * 16 + keyAdd + r;
                        st[f][r] += (key > qcol) ? -1e9f : 0.f;
                    }
            }
            float mNew = mI;
#pragma unroll
            for (int f = 0; f < 4; ++f)
#pragma unroll
                for (int r = 0; r < 8; ++r) mNew = fmaxf(mNew, st[f][r]);
            mNew = fmaxf(mNew, xor16(mNew));
            float alpha = __expf(mI - mNew);
            mI = mNew;
            float ls = 0.f;
#pragma unroll
            for (int f = 0; f < 4; ++f)
#pragma unroll
                for (int r = 0; r < 8; ++r) {
                    float p = __expf(st[f][r] - mNew);
                    st[f][r] = p;
                    ls += p;
                }
            ls += xor16(ls);
            lI = lI * alpha + ls;
#pragma unroll
            for (int j = 0; j < 4; ++j)
#pragma unroll
                for (int r = 0; r < 8; ++r) o[j][r] *= alpha;
            // repack P^T (C layout) in-lane into B-fragment layout
            u32x8 up0, up1;
#pragma unroll
            for (int i = 0; i < 4; ++i) {
                up0[i]     = cvt_pk_bf16(st[0][2 * i], st[0][2 * i + 1]);
                up0[4 + i] = cvt_pk_bf16(st[1][2 * i], st[1][2 * i + 1]);
                up1[i]     = cvt_pk_bf16(st[2][2 * i], st[2][2 * i + 1]);
                up1[4 + i] = cvt_pk_bf16(st[3][2 * i], st[3][2 * i + 1]);
            }
            bf16x16 pb0 = __builtin_bit_cast(bf16x16, up0);
            bf16x16 pb1 = __builtin_bit_cast(bf16x16, up1);
            // O^T += V^T x P^T  (A = Vs rows = depth, k = keys)
#pragma unroll
            for (int j = 0; j < 4; ++j) {
                bf16x16 v0 = ldfrag(&Vs[cur][(j * 16) * LDSW], LDSW, 0, lane);
                bf16x16 v1 = ldfrag(&Vs[cur][(j * 16) * LDSW], LDSW, 32, lane);
                o[j] = wmma_bf16(v0, pb0, o[j]);
                o[j] = wmma_bf16(v1, pb1, o[j]);
            }
        }
        __syncthreads();
    }

    const float inv = 1.f / lI;
    const int s_ = qRow0 + (lane & 15);
    __bf16* orow = Out + ((size_t)(b * SEQ + s_)) * D_MODEL + h * DEPTH;
#pragma unroll
    for (int j = 0; j < 4; ++j)
#pragma unroll
        for (int r = 0; r < 8; r += 2) {
            int d = j * 16 + keyAdd + r;     // even -> dword aligned
            unsigned pr = cvt_pk_bf16(o[j][r] * inv, o[j][r + 1] * inv);
            *reinterpret_cast<unsigned*>(orow + d) = pr;
        }
}

// ---------------------------------------------------------------------------
extern "C" void kernel_launch(void* const* d_in, const int* in_sizes, int n_in,
                              void* d_out, int out_size, void* d_ws, size_t ws_size,
                              hipStream_t stream) {
    const float* q  = (const float*)d_in[0];
    const float* k  = (const float*)d_in[1];
    const float* v  = (const float*)d_in[2];
    // d_in[3] = mask (causal, applied analytically)
    const float* Wq = (const float*)d_in[4];
    const float* bq = (const float*)d_in[5];
    const float* Wk = (const float*)d_in[6];
    const float* bk = (const float*)d_in[7];
    const float* Wv = (const float*)d_in[8];
    const float* bv = (const float*)d_in[9];
    const float* Wo = (const float*)d_in[10];
    const float* bo = (const float*)d_in[11];

    char* ws = (char*)d_ws;
    const size_t MB = 1024ull * 1024ull;
    __bf16* Wtq = (__bf16*)(ws + 0 * MB);
    __bf16* Wtk = (__bf16*)(ws + 2 * MB);
    __bf16* Wtv = (__bf16*)(ws + 4 * MB);
    __bf16* Wto = (__bf16*)(ws + 6 * MB);
    __bf16* Qbf = (__bf16*)(ws + 8 * MB);     // [B,H,S,64]  16MB
    __bf16* Kbf = (__bf16*)(ws + 24 * MB);    // [B,H,S,64]  16MB
    __bf16* Vtb = (__bf16*)(ws + 40 * MB);    // [B,H,64,S]  16MB
    __bf16* Abf = (__bf16*)(ws + 56 * MB);    // [B,S,1024]  16MB

    dim3 tgrid((D_MODEL * D_MODEL) / 256);
    wtr_73091753443647_kernel<<<tgrid, 256, 0, stream>>>(Wq, Wtq);
    wtr_73091753443647_kernel<<<tgrid, 256, 0, stream>>>(Wk, Wtk);
    wtr_73091753443647_kernel<<<tgrid, 256, 0, stream>>>(Wv, Wtv);
    wtr_73091753443647_kernel<<<tgrid, 256, 0, stream>>>(Wo, Wto);

    dim3 ggrid(D_MODEL / 128, MTOT / 128);    // (8, 64)
    gemm_73091753443647_kernel<0, false><<<ggrid, 256, 0, stream>>>(q, Wtq, bq, Qbf, 0.125f);
    gemm_73091753443647_kernel<0, false><<<ggrid, 256, 0, stream>>>(k, Wtk, bk, Kbf, 1.0f);
    gemm_73091753443647_kernel<1, false><<<ggrid, 256, 0, stream>>>(v, Wtv, bv, Vtb, 1.0f);

    dim3 agrid(SEQ / 128, NUM_HEADS, BATCH);  // (16,16,4)
    attn_73091753443647_kernel<<<agrid, 256, 0, stream>>>(Qbf, Kbf, Vtb, Abf);

    gemm_73091753443647_kernel<2, true><<<ggrid, 256, 0, stream>>>(Abf, Wto, bo, d_out, 1.0f);
}
